// GraphTransformerLayer_82995948028015
// MI455X (gfx1250) — matrix operand order repaired and checked by reading, not compiled, not run on hardware
//
#include <hip/hip_runtime.h>
#include <math.h>

#define NN 50000
#define NE 600000
#define DD 128
#define HH 8
#define DHH 16
#define ATTN_SCALE 0.08838834764831845f  // 128^-0.5

typedef __attribute__((ext_vector_type(16))) __bf16 v16bf;
typedef __attribute__((ext_vector_type(8)))  float v8f;

// ---------- bf16 helpers ----------
static __device__ __forceinline__ unsigned short f2bf(float f) {
    unsigned u = __float_as_uint(f);
    unsigned r = 0x7FFFu + ((u >> 16) & 1u);
    return (unsigned short)((u + r) >> 16);
}
static __device__ __forceinline__ float bf2f(unsigned short h) {
    return __uint_as_float(((unsigned)h) << 16);
}
static __device__ __forceinline__ void split_bf(float x, unsigned short& hi, unsigned short& lo) {
    hi = f2bf(x);
    lo = f2bf(x - bf2f(hi));
}

// ---------- orderable-uint encoding for float atomic max ----------
static __device__ __forceinline__ unsigned encodef(float f) {
    unsigned u = __float_as_uint(f);
    return (u & 0x80000000u) ? ~u : (u | 0x80000000u);
}
static __device__ __forceinline__ float decodef(unsigned e) {
    return (e & 0x80000000u) ? __uint_as_float(e & 0x7FFFFFFFu) : __uint_as_float(~e);
}

// ---------- WMMA fragment containers ----------
union BFrag16 {
    unsigned short u[16];
    unsigned int   d[8];
    uint4          q[2];
    v16bf          v;
};

static __device__ __forceinline__ v8f wmma_bf16(v16bf a, v16bf b, v8f c) {
    return __builtin_amdgcn_wmma_f32_16x16x32_bf16(false, a, false, b, (short)0, c, false, false);
}

// A-matrix (16x32 bf16) per ISA 7.12.2: lanes 0-15 rows M=0..15 (K base 0),
// lanes 16-31 same rows, K offset +8; VGPR v<4 -> K=2v(+off), v>=4 -> K=16+2(v-4)(+off).
// The two 4-VGPR groups are each 16 contiguous bytes -> two b128 loads.
static __device__ __forceinline__ void load_a_frag(BFrag16& f, const unsigned short* A,
                                                   int lda, int row0, int k0, int lane) {
    int half = lane >> 4;
    int m    = lane & 15;
    const unsigned short* base = A + (size_t)(row0 + m) * lda + k0 + half * 8;
    f.q[0] = *(const uint4*)(base);        // K = k0+off .. +7   (pairs in VGPRs 0..3)
    f.q[1] = *(const uint4*)(base + 16);   // K = k0+off+16 .. +23 (VGPRs 4..7)
}

// B-matrix (32x16 bf16): lane holds column n=lane%16; u[0..7] K=8h..8h+7, u[8..15]
// K=16+8h..16+8h+7 (h=lane/16, same K order as A). Weights are stored pre-TRANSPOSED
// (WT[Nc][K]) so each 8-K group is 16 contiguous bytes -> two b128 loads.
static __device__ __forceinline__ void load_b_fragT(BFrag16& f, const unsigned short* WT,
                                                    int ldk, int k0, int n0, int lane) {
    int half = lane >> 4;
    int n    = lane & 15;
    const unsigned short* base = WT + (size_t)(n0 + n) * ldk + k0 + half * 8;
    f.q[0] = *(const uint4*)(base);
    f.q[1] = *(const uint4*)(base + 16);
}

// ---------- utility kernels ----------
__global__ void zero_f32(float* p, long long n) {
    long long i = (long long)blockIdx.x * blockDim.x + threadIdx.x;
    if (i < n) p[i] = 0.0f;
}

__global__ void split_f32_to_bf(const float* __restrict__ src,
                                unsigned short* __restrict__ hi,
                                unsigned short* __restrict__ lo, long long n) {
    long long i = (long long)blockIdx.x * blockDim.x + threadIdx.x;
    if (i < n) {
        unsigned short h, l;
        split_bf(src[i], h, l);
        hi[i] = h; lo[i] = l;
    }
}

// split + transpose: W[K][Nc] f32 -> WT_hi/WT_lo[Nc][K] bf16
__global__ void split_transpose_bf(const float* __restrict__ src,
                                   unsigned short* __restrict__ hiT,
                                   unsigned short* __restrict__ loT, int K, int Nc) {
    int i = blockIdx.x * blockDim.x + threadIdx.x;
    if (i >= K * Nc) return;
    int k = i / Nc, n = i % Nc;
    unsigned short h, l;
    split_bf(src[i], h, l);
    hiT[(size_t)n * K + k] = h;
    loT[(size_t)n * K + k] = l;
}

// ---------- LayerNorm: one wave32 per row of 128 ----------
__global__ __launch_bounds__(256) void layernorm_kernel(
    const float* __restrict__ x, const float* __restrict__ g, const float* __restrict__ b,
    float* __restrict__ outf,                 // may be null
    unsigned short* __restrict__ out_hi, unsigned short* __restrict__ out_lo,
    int rows) {
    int lane = threadIdx.x & 31;
    int row  = blockIdx.x * (blockDim.x >> 5) + (threadIdx.x >> 5);
    if (row >= rows) return;
    const float4* xr = (const float4*)(x + (size_t)row * DD);
    float4 v = xr[lane];
    float s  = v.x + v.y + v.z + v.w;
    float sq = v.x * v.x + v.y * v.y + v.z * v.z + v.w * v.w;
#pragma unroll
    for (int off = 16; off > 0; off >>= 1) {
        s  += __shfl_xor(s,  off, 32);
        sq += __shfl_xor(sq, off, 32);
    }
    float mean = s * (1.0f / DD);
    float var  = sq * (1.0f / DD) - mean * mean;
    float rstd = rsqrtf(var + 1e-5f);
    float4 gg = ((const float4*)g)[lane];
    float4 bb = ((const float4*)b)[lane];
    float e[4]  = {v.x, v.y, v.z, v.w};
    float gv[4] = {gg.x, gg.y, gg.z, gg.w};
    float bv[4] = {bb.x, bb.y, bb.z, bb.w};
#pragma unroll
    for (int j = 0; j < 4; ++j) {
        int idx = lane * 4 + j;
        float y = (e[j] - mean) * rstd * gv[j] + bv[j];
        if (outf) outf[(size_t)row * DD + idx] = y;
        unsigned short h, l;
        split_bf(y, h, l);
        out_hi[(size_t)row * DD + idx] = h;
        out_lo[(size_t)row * DD + idx] = l;
    }
}

// ---------- split-bf16 GEMM: C[M,Nc] = A[M,K] @ W[K,Nc] + bias (+epilogue) ----------
// Weights come in pre-transposed (WT[Nc][K]). K is compile-time so the 4-step
// k-loop fully unrolls: loads cluster ahead of a 12-deep WMMA chain.
// MODE 0: scale columns < 128 by ATTN_SCALE (QKV q-scaling); MODE 1: add residual.
template <int K, int MODE>
__global__ __launch_bounds__(256) void gemm_split_bf16(
    const unsigned short* __restrict__ Ahi, const unsigned short* __restrict__ Alo,
    const unsigned short* __restrict__ BhiT, const unsigned short* __restrict__ BloT,
    const float* __restrict__ bias, const float* __restrict__ residual,
    float* __restrict__ C, int M, int Nc) {
    int lane   = threadIdx.x & 31;
    int wid    = blockIdx.x * (blockDim.x >> 5) + (threadIdx.x >> 5);
    int tilesN = Nc >> 4;
    int tileM  = wid / tilesN;
    int tileN  = wid % tilesN;
    if (tileM * 16 >= M) return;
    int row0 = tileM * 16, col0 = tileN * 16;

    v8f acc = {0.f, 0.f, 0.f, 0.f, 0.f, 0.f, 0.f, 0.f};
#pragma unroll
    for (int k0 = 0; k0 < K; k0 += 32) {
        BFrag16 ah, al, bh, bl;
        load_a_frag(ah, Ahi, K, row0, k0, lane);
        load_a_frag(al, Alo, K, row0, k0, lane);
        load_b_fragT(bh, BhiT, K, k0, col0, lane);
        load_b_fragT(bl, BloT, K, k0, col0, lane);
        acc = wmma_bf16(ah.v, bh.v, acc);
        acc = wmma_bf16(ah.v, bl.v, acc);
        acc = wmma_bf16(al.v, bh.v, acc);
    }
    int half = lane >> 4;
    int n    = lane & 15;
    int col  = col0 + n;
    float bv = bias[col];
#pragma unroll
    for (int v = 0; v < 8; ++v) {
        int row   = row0 + v + half * 8;
        float val = acc[v] + bv;
        if (MODE == 0) { if (col < 128) val *= ATTN_SCALE; }
        else           { val += residual[(size_t)row * Nc + col]; }
        C[(size_t)row * Nc + col] = val;
    }
}

// ---------- edge pass 1: scores + segment max (atomic on ordered uints) ----------
__global__ void edge_pass1(const float* __restrict__ qkv, const int* __restrict__ src,
                           const int* __restrict__ dst, const float* __restrict__ dist,
                           const float* __restrict__ path, float* __restrict__ exbuf,
                           unsigned* __restrict__ mxu) {
    long long tid = (long long)blockIdx.x * blockDim.x + threadIdx.x;
    if (tid >= (long long)NE * HH) return;
    int e = (int)(tid >> 3), h = (int)(tid & 7);
    int s = src[e], d = dst[e];
    const float4* qp = (const float4*)(qkv + (size_t)s * 384 + h * DHH);
    const float4* kp = (const float4*)(qkv + (size_t)d * 384 + 128 + h * DHH);
    float dot = 0.0f;
#pragma unroll
    for (int i = 0; i < 4; ++i) {
        float4 a = qp[i], b = kp[i];
        dot += a.x * b.x + a.y * b.y + a.z * b.z + a.w * b.w;
    }
    float a_val = dot + dist[(size_t)e * HH + h] + path[(size_t)e * HH + h];
    exbuf[tid] = a_val;
    atomicMax(&mxu[(size_t)d * HH + h], encodef(a_val));
}

// ---------- edge pass 2: exp(a - max) + segment sum ----------
__global__ void edge_pass2(const int* __restrict__ dst, const unsigned* __restrict__ mxu,
                           float* __restrict__ exbuf, float* __restrict__ den) {
    long long tid = (long long)blockIdx.x * blockDim.x + threadIdx.x;
    if (tid >= (long long)NE * HH) return;
    int e = (int)(tid >> 3), h = (int)(tid & 7);
    int d = dst[e];
    float mx = decodef(mxu[(size_t)d * HH + h]);
    float ex = expf(exbuf[tid] - mx);
    exbuf[tid] = ex;
    atomicAdd(&den[(size_t)d * HH + h], ex);
}

// ---------- edge pass 3: weighted v scatter-add ----------
__global__ void edge_pass3(const float* __restrict__ qkv, const int* __restrict__ src,
                           const int* __restrict__ dst, const float* __restrict__ exbuf,
                           const float* __restrict__ den, float* __restrict__ agg) {
    long long tid = (long long)blockIdx.x * blockDim.x + threadIdx.x;
    if (tid >= (long long)NE * DD) return;
    int e = (int)(tid >> 7), dcol = (int)(tid & 127), h = dcol >> 4;
    int s = src[e], d = dst[e];
    float sa  = exbuf[(size_t)e * HH + h] / den[(size_t)d * HH + h];
    float val = qkv[(size_t)s * 384 + 256 + dcol] * sa;
    atomicAdd(&agg[(size_t)d * DD + dcol], val);
}

// ---------- fused MLP: gelu(y0 @ W1 + b1) @ W2 + b2, + x residual -> out ----------
// GELU intermediate (16x512 bf16 hi/lo) lives in LDS; row stride 520 elems
// (1040 B = 65*16 B) keeps b128 alignment and spreads rows across banks.
#define LDPAD 520
__global__ __launch_bounds__(256) void mlp_fused(
    const unsigned short* __restrict__ y0hi, const unsigned short* __restrict__ y0lo,
    const unsigned short* __restrict__ w1hiT, const unsigned short* __restrict__ w1loT,
    const float* __restrict__ b1,
    const unsigned short* __restrict__ w2hiT, const unsigned short* __restrict__ w2loT,
    const float* __restrict__ b2,
    const float* __restrict__ x, float* __restrict__ out) {
    __shared__ unsigned short t_hi[16][LDPAD];
    __shared__ unsigned short t_lo[16][LDPAD];

    int lane = threadIdx.x & 31;
    int w    = threadIdx.x >> 5;
    int row0 = blockIdx.x * 16;
    int half = lane >> 4;
    int n    = lane & 15;

    // Stage 1: t = gelu(y0 @ W1 + b1), each wave covers 4 of 32 column tiles.
#pragma unroll
    for (int i = 0; i < 4; ++i) {
        int col0 = (w * 4 + i) * 16;
        v8f acc  = {0.f, 0.f, 0.f, 0.f, 0.f, 0.f, 0.f, 0.f};
#pragma unroll
        for (int k0 = 0; k0 < 128; k0 += 32) {
            BFrag16 ah, al, bh, bl;
            load_a_frag(ah, y0hi, 128, row0, k0, lane);
            load_a_frag(al, y0lo, 128, row0, k0, lane);
            load_b_fragT(bh, w1hiT, 128, k0, col0, lane);
            load_b_fragT(bl, w1loT, 128, k0, col0, lane);
            acc = wmma_bf16(ah.v, bh.v, acc);
            acc = wmma_bf16(ah.v, bl.v, acc);
            acc = wmma_bf16(al.v, bh.v, acc);
        }
        int col  = col0 + n;
        float bb = b1[col];
#pragma unroll
        for (int v = 0; v < 8; ++v) {
            int m     = v + half * 8;
            float val = acc[v] + bb;
            // tanh-approx GELU (jax.nn.gelu default)
            float g = 0.5f * val *
                      (1.0f + tanhf(0.7978845608028654f * (val + 0.044715f * val * val * val)));
            unsigned short h, l;
            split_bf(g, h, l);
            t_hi[m][col] = h;
            t_lo[m][col] = l;
        }
    }
    __syncthreads();

    // Stage 2: out = x + t @ W2 + b2, one 16x16 tile per wave (8 waves, 128 cols).
    int col0 = w * 16;
    v8f acc  = {0.f, 0.f, 0.f, 0.f, 0.f, 0.f, 0.f, 0.f};
#pragma unroll 4
    for (int k0 = 0; k0 < 512; k0 += 32) {
        BFrag16 ah, al, bh, bl;
        load_a_frag(ah, &t_hi[0][0], LDPAD, 0, k0, lane);
        load_a_frag(al, &t_lo[0][0], LDPAD, 0, k0, lane);
        load_b_fragT(bh, w2hiT, 512, k0, col0, lane);
        load_b_fragT(bl, w2loT, 512, k0, col0, lane);
        acc = wmma_bf16(ah.v, bh.v, acc);
        acc = wmma_bf16(ah.v, bl.v, acc);
        acc = wmma_bf16(al.v, bh.v, acc);
    }
    int col  = col0 + n;
    float bb = b2[col];
#pragma unroll
    for (int v = 0; v < 8; ++v) {
        int row    = row0 + v + half * 8;
        size_t idx = (size_t)row * DD + col;
        out[idx]   = x[idx] + acc[v] + bb;
    }
}

// ---------- host launcher ----------
extern "C" void kernel_launch(void* const* d_in, const int* in_sizes, int n_in,
                              void* d_out, int out_size, void* d_ws, size_t ws_size,
                              hipStream_t stream) {
    const float* node_feature = (const float*)d_in[0];
    const float* dist_attn    = (const float*)d_in[1];
    const float* path_attn    = (const float*)d_in[2];
    const int*   src          = (const int*)d_in[3];
    const int*   dst          = (const int*)d_in[4];
    const float* ln1_g        = (const float*)d_in[5];
    const float* ln1_b        = (const float*)d_in[6];
    const float* qkv_w        = (const float*)d_in[7];
    const float* qkv_b        = (const float*)d_in[8];
    const float* in_proj_w    = (const float*)d_in[9];
    const float* in_proj_b    = (const float*)d_in[10];
    const float* res_ln_g     = (const float*)d_in[11];
    const float* res_ln_b     = (const float*)d_in[12];
    const float* mlp_w1       = (const float*)d_in[13];
    const float* mlp_b1       = (const float*)d_in[14];
    const float* mlp_w2       = (const float*)d_in[15];
    const float* mlp_b2       = (const float*)d_in[16];
    float* out = (float*)d_out;

    // workspace carve-up (256B aligned)
    char* base = (char*)d_ws;
    size_t off = 0;
    auto alloc = [&](size_t bytes) -> char* {
        char* p = base + off;
        off = (off + bytes + 255) & ~(size_t)255;
        return p;
    };
    float*          h_f32  = (float*)alloc((size_t)NN * DD * 4);
    unsigned short* h_hi   = (unsigned short*)alloc((size_t)NN * DD * 2);
    unsigned short* h_lo   = (unsigned short*)alloc((size_t)NN * DD * 2);
    float*          qkv    = (float*)alloc((size_t)NN * 384 * 4);
    float*          agg    = (float*)alloc((size_t)NN * DD * 4);
    unsigned short* agg_hi = (unsigned short*)alloc((size_t)NN * DD * 2);
    unsigned short* agg_lo = (unsigned short*)alloc((size_t)NN * DD * 2);
    float*          xbuf   = (float*)alloc((size_t)NN * DD * 4);
    unsigned short* y0_hi  = (unsigned short*)alloc((size_t)NN * DD * 2);
    unsigned short* y0_lo  = (unsigned short*)alloc((size_t)NN * DD * 2);
    unsigned*       mxu    = (unsigned*)alloc((size_t)NN * HH * 4);
    float*          den    = (float*)alloc((size_t)NN * HH * 4);
    float*          exbuf  = (float*)alloc((size_t)NE * HH * 4);
    unsigned short* qw_hiT = (unsigned short*)alloc(128 * 384 * 2);
    unsigned short* qw_loT = (unsigned short*)alloc(128 * 384 * 2);
    unsigned short* pw_hiT = (unsigned short*)alloc(128 * 128 * 2);
    unsigned short* pw_loT = (unsigned short*)alloc(128 * 128 * 2);
    unsigned short* w1_hiT = (unsigned short*)alloc(128 * 512 * 2);
    unsigned short* w1_loT = (unsigned short*)alloc(128 * 512 * 2);
    unsigned short* w2_hiT = (unsigned short*)alloc(512 * 128 * 2);
    unsigned short* w2_loT = (unsigned short*)alloc(512 * 128 * 2);
    (void)ws_size; (void)in_sizes; (void)n_in; (void)out_size;

    auto cdiv = [](long long a, long long b) { return (int)((a + b - 1) / b); };

    // weight hi/lo split + transpose (tiny; redone each call for determinism)
    split_transpose_bf<<<cdiv(128 * 384, 256), 256, 0, stream>>>(qkv_w, qw_hiT, qw_loT, 128, 384);
    split_transpose_bf<<<cdiv(128 * 128, 256), 256, 0, stream>>>(in_proj_w, pw_hiT, pw_loT, 128, 128);
    split_transpose_bf<<<cdiv(128 * 512, 256), 256, 0, stream>>>(mlp_w1, w1_hiT, w1_loT, 128, 512);
    split_transpose_bf<<<cdiv(512 * 128, 256), 256, 0, stream>>>(mlp_w2, w2_hiT, w2_loT, 512, 128);

    // LN1
    layernorm_kernel<<<cdiv(NN, 8), 256, 0, stream>>>(node_feature, ln1_g, ln1_b,
                                                      h_f32, h_hi, h_lo, NN);

    // QKV GEMM: 3125 x 24 tiles, 8 waves/block
    gemm_split_bf16<128, 0><<<(3125 * 24) / 8, 256, 0, stream>>>(h_hi, h_lo, qw_hiT, qw_loT,
                                                                 qkv_b, nullptr, qkv, NN, 384);

    // init accumulators (mxu zero == encoded minimum)
    zero_f32<<<cdiv((long long)NN * DD, 256), 256, 0, stream>>>(agg, (long long)NN * DD);
    zero_f32<<<cdiv((long long)NN * HH, 256), 256, 0, stream>>>(den, (long long)NN * HH);
    zero_f32<<<cdiv((long long)NN * HH, 256), 256, 0, stream>>>((float*)mxu, (long long)NN * HH);

    // edge attention (segment softmax over dst)
    edge_pass1<<<cdiv((long long)NE * HH, 256), 256, 0, stream>>>(qkv, src, dst, dist_attn,
                                                                  path_attn, exbuf, mxu);
    edge_pass2<<<cdiv((long long)NE * HH, 256), 256, 0, stream>>>(dst, mxu, exbuf, den);
    edge_pass3<<<cdiv((long long)NE * DD, 256), 256, 0, stream>>>(qkv, src, dst, exbuf, den, agg);

    // split agg for WMMA
    split_f32_to_bf<<<cdiv((long long)NN * DD, 256), 256, 0, stream>>>(agg, agg_hi, agg_lo,
                                                                       (long long)NN * DD);

    // x = h + agg @ in_proj_w + b
    gemm_split_bf16<128, 1><<<(3125 * 8) / 8, 256, 0, stream>>>(agg_hi, agg_lo, pw_hiT, pw_loT,
                                                                in_proj_b, h_f32, xbuf, NN, 128);

    // LN2 (bf16 splits only)
    layernorm_kernel<<<cdiv(NN, 8), 256, 0, stream>>>(xbuf, res_ln_g, res_ln_b,
                                                      nullptr, y0_hi, y0_lo, NN);

    // fused MLP + final residual
    mlp_fused<<<3125, 256, 0, stream>>>(y0_hi, y0_lo, w1_hiT, w1_loT, mlp_b1,
                                        w2_hiT, w2_loT, mlp_b2, xbuf, out);
}
